// GCL_9251359556274
// MI455X (gfx1250) — matrix operand order repaired and checked by reading, not compiled, not run on hardware
//
#include <hip/hip_runtime.h>
#include <hip/hip_bf16.h>

typedef __attribute__((ext_vector_type(16))) _Float16 v16h;
typedef __attribute__((ext_vector_type(8)))  _Float16 v8h;
typedef __attribute__((ext_vector_type(8)))  float    v8f;

#define TPB 256

// ---------------- degree / normalization ----------------

__global__ void k_init_deg(float* __restrict__ deg, int n) {
    int i = blockIdx.x * blockDim.x + threadIdx.x;
    if (i < n) deg[i] = 1.0f;   // self-loop contributes 1
}

__global__ void k_count_deg(const int* __restrict__ dst, float* __restrict__ deg, int e) {
    int i = blockIdx.x * blockDim.x + threadIdx.x;
    if (i < e) atomicAdd(&deg[dst[i]], 1.0f);
}

__global__ void k_to_dis(float* __restrict__ deg, int n) {
    int i = blockIdx.x * blockDim.x + threadIdx.x;
    if (i < n) deg[i] = rsqrtf(fmaxf(deg[i], 1.0f));  // deg -> dis in place
}

// ---------------- conversions / utility ----------------

__global__ void k_f32_to_f16(const float* __restrict__ in, _Float16* __restrict__ out, long long n) {
    long long i = (long long)blockIdx.x * blockDim.x + threadIdx.x;
    if (i < n) out[i] = (_Float16)in[i];
}

__global__ void k_zero_f32(float* __restrict__ p, long long n) {
    long long i = (long long)blockIdx.x * blockDim.x + threadIdx.x;
    if (i < n) p[i] = 0.0f;
}

// -------- pack W[K x Nc] (f32, row-major) into WMMA B-fragment order --------
// Packed layout: Bpk[((ktile*tilesN + tileN)*32 + lane)*16 + j]
//   lane<16 : column tileN*16+lane,    K = ktile*32 + {0..7, 16..23}[j]  (j=0..15)
//   lane>=16: column tileN*16+lane-16, K = ktile*32 + {8..15, 24..31}[j]
// => in the GEMM each lane reads its whole 32-byte fragment contiguously.

__global__ void k_pack_B(const float* __restrict__ W, _Float16* __restrict__ Bpk,
                         int K, int Nc) {
    int t = blockIdx.x * blockDim.x + threadIdx.x;
    if (t >= K * Nc) return;
    const int j      = t & 15;
    const int lane   = (t >> 4) & 31;
    const int tile   = t >> 9;          // ktile*tilesN + tileN
    const int tilesN = Nc >> 4;
    const int ktile  = tile / tilesN;
    const int tileN  = tile % tilesN;
    const int half   = lane >> 4;
    const int l15    = lane & 15;
    const int k      = (ktile << 5) + (half << 3) + (j & 7) + ((j >> 3) << 4);
    const int c      = (tileN << 4) + l15;
    Bpk[t] = (_Float16)W[(long long)k * Nc + c];
}

// ---------------- WMMA GEMM: C[MxNc] = A[MxK] @ Bpk, f16 in, f32 out --------
// One wave per 16x16 output tile; K fully unrolled (template).
// Guard is wave-uniform -> EXEC all-1s at every WMMA (ISA requirement).

template <int K>
__global__ void k_gemm_wmma_f16(const _Float16* __restrict__ A,
                                const _Float16* __restrict__ Bpk,
                                float* __restrict__ C,
                                int Nc, long long totalTiles, int tilesN) {
    const int waveInBlk = threadIdx.x >> 5;
    const int lane      = threadIdx.x & 31;
    const long long wid = (long long)blockIdx.x * (blockDim.x >> 5) + waveInBlk;
    if (wid >= totalTiles) return;              // uniform across the wave

    const int tileM = (int)(wid / tilesN);
    const int tileN = (int)(wid % tilesN);
    const int row0  = tileM << 4;
    const int col0  = tileN << 4;

    const int half = lane >> 4;                 // 0: lanes 0-15, 1: lanes 16-31
    const int l15  = lane & 15;
    const int mrow = row0 + l15;                // A row for this lane
    const int bcol = col0 + l15;                // C column for this lane

    const _Float16* arow = A + (long long)mrow * K + half * 8;
    const _Float16* brow = Bpk + ((long long)tileN << 9) + (lane << 4);

    v8f acc = {};
    #pragma unroll
    for (int k0 = 0; k0 < K; k0 += 32) {
        // A fragment (16x32 f16): half=0 -> K {k0..k0+7, k0+16..k0+23};
        //                         half=1 -> K {k0+8..k0+15, k0+24..k0+31}
        const v8h alo = *(const v8h*)(arow + k0);        // 16B aligned
        const v8h ahi = *(const v8h*)(arow + k0 + 16);   // 16B aligned
        v16h a;
        #pragma unroll
        for (int j = 0; j < 8; ++j) { a[j] = alo[j]; a[8 + j] = ahi[j]; }

        // B fragment: one contiguous 32B load from packed weights
        const v16h b = *(const v16h*)(brow + ((long long)(k0 >> 5) * tilesN << 9));

        acc = __builtin_amdgcn_wmma_f32_16x16x32_f16(
            /*neg_a=*/false, a, /*neg_b=*/false, b,
            /*c_mod=*/(short)0, acc, /*reuse_a=*/false, /*reuse_b=*/false);
    }

    // C/D layout: VGPR v -> row row0 + half*8 + v, column bcol
    const int mb = half << 3;
    #pragma unroll
    for (int v = 0; v < 8; ++v)
        C[(long long)(row0 + mb + v) * Nc + bcol] = acc[v];
}

// runtime-K fallback (not expected to be used for this problem's shapes)
__global__ void k_gemm_wmma_f16_dyn(const _Float16* __restrict__ A,
                                    const _Float16* __restrict__ Bpk,
                                    float* __restrict__ C,
                                    int K, int Nc, long long totalTiles, int tilesN) {
    const int waveInBlk = threadIdx.x >> 5;
    const int lane      = threadIdx.x & 31;
    const long long wid = (long long)blockIdx.x * (blockDim.x >> 5) + waveInBlk;
    if (wid >= totalTiles) return;
    const int tileM = (int)(wid / tilesN);
    const int tileN = (int)(wid % tilesN);
    const int row0 = tileM << 4, col0 = tileN << 4;
    const int half = lane >> 4, l15 = lane & 15;
    const _Float16* arow = A + (long long)(row0 + l15) * K + half * 8;
    const _Float16* brow = Bpk + ((long long)tileN << 9) + (lane << 4);
    v8f acc = {};
    for (int k0 = 0; k0 < K; k0 += 32) {
        const v8h alo = *(const v8h*)(arow + k0);
        const v8h ahi = *(const v8h*)(arow + k0 + 16);
        v16h a;
        #pragma unroll
        for (int j = 0; j < 8; ++j) { a[j] = alo[j]; a[8 + j] = ahi[j]; }
        const v16h b = *(const v16h*)(brow + ((long long)(k0 >> 5) * tilesN << 9));
        acc = __builtin_amdgcn_wmma_f32_16x16x32_f16(false, a, false, b, (short)0, acc, false, false);
    }
    const int mb = half << 3;
    #pragma unroll
    for (int v = 0; v < 8; ++v)
        C[(long long)(row0 + mb + v) * Nc + (col0 + l15)] = acc[v];
}

// ---------------- edge scatter: agg[dst] += hw[src] * dis[src]*dis[dst] ----

__global__ void k_scatter(const int* __restrict__ src, const int* __restrict__ dst,
                          const float* __restrict__ dis,
                          const float* __restrict__ hw, float* __restrict__ agg,
                          int E, int Fq /*F/4*/, int F) {
    long long t = (long long)blockIdx.x * blockDim.x + threadIdx.x;
    if (t >= (long long)E * Fq) return;
    const int e = (int)(t / Fq);
    const int c = (int)(t % Fq) << 2;
    const int s = src[e], d = dst[e];
    const float nrm = dis[s] * dis[d];
    const float4 v = *(const float4*)(hw + (long long)s * F + c);
    float* base = agg + (long long)d * F + c;
    atomicAdd(base + 0, v.x * nrm);
    atomicAdd(base + 1, v.y * nrm);
    atomicAdd(base + 2, v.z * nrm);
    atomicAdd(base + 3, v.w * nrm);
}

// ------------- epilogue layer1: h1 = relu(agg + hw*dis^2 + b), -> f16 -------

__global__ void k_post_relu(const float* __restrict__ agg, const float* __restrict__ hw,
                            const float* __restrict__ dis, const float* __restrict__ bias,
                            _Float16* __restrict__ out16, int n, int F) {
    long long t = (long long)blockIdx.x * blockDim.x + threadIdx.x;
    if (t >= (long long)n * F) return;
    const int i = (int)(t / F);
    const int f = (int)(t % F);
    const float di = dis[i];
    float v = agg[t] + hw[t] * di * di + bias[f];
    v = fmaxf(v, 0.0f);
    out16[t] = (_Float16)v;
}

// ------ epilogue layer2: h2 = agg + hw*dis^2 + b, -> f32 (d_out) + f16 ------

__global__ void k_post(const float* __restrict__ agg, const float* __restrict__ hw,
                       const float* __restrict__ dis, const float* __restrict__ bias,
                       float* __restrict__ outf, _Float16* __restrict__ out16,
                       int n, int F) {
    long long t = (long long)blockIdx.x * blockDim.x + threadIdx.x;
    if (t >= (long long)n * F) return;
    const int i = (int)(t / F);
    const int f = (int)(t % F);
    const float di = dis[i];
    const float v = agg[t] + hw[t] * di * di + bias[f];
    outf[t]  = v;
    out16[t] = (_Float16)v;
}

// ---------------- classifier epilogue: out = hw + bc ----------------

__global__ void k_add_bias(const float* __restrict__ hw, const float* __restrict__ bias,
                           float* __restrict__ out, long long total, int F) {
    long long t = (long long)blockIdx.x * blockDim.x + threadIdx.x;
    if (t >= total) return;
    out[t] = hw[t] + bias[(int)(t % F)];
}

// ---------------------------------------------------------------------------

static inline unsigned gridFor(long long n) { return (unsigned)((n + TPB - 1) / TPB); }

static void launch_gemm(const _Float16* A, const _Float16* Bpk, float* C,
                        int M, int K, int Nc, hipStream_t stream) {
    const long long tiles  = (long long)(M / 16) * (Nc / 16);
    const unsigned  blocks = (unsigned)((tiles * 32 + 127) / 128);
    const int tilesN = Nc / 16;
    if (K == 128)
        k_gemm_wmma_f16<128><<<blocks, 128, 0, stream>>>(A, Bpk, C, Nc, tiles, tilesN);
    else if (K == 64)
        k_gemm_wmma_f16<64><<<blocks, 128, 0, stream>>>(A, Bpk, C, Nc, tiles, tilesN);
    else
        k_gemm_wmma_f16_dyn<<<blocks, 128, 0, stream>>>(A, Bpk, C, K, Nc, tiles, tilesN);
}

extern "C" void kernel_launch(void* const* d_in, const int* in_sizes, int n_in,
                              void* d_out, int out_size, void* d_ws, size_t ws_size,
                              hipStream_t stream) {
    const float* x  = (const float*)d_in[0];
    const int*   ei = (const int*)d_in[1];
    const float* W1 = (const float*)d_in[2];
    const float* b1 = (const float*)d_in[3];
    const float* W2 = (const float*)d_in[4];
    const float* b2 = (const float*)d_in[5];
    const float* Wc = (const float*)d_in[6];
    const float* bc = (const float*)d_in[7];
    float* out = (float*)d_out;

    const int HID = in_sizes[3];               // 128
    const int OUT = in_sizes[5];               // 64
    const int DIN = in_sizes[2] / HID;         // 128
    const int N   = in_sizes[0] / DIN;         // 50000
    const int E   = in_sizes[1] / 2;           // 800000
    const int* src = ei;
    const int* dst = ei + E;

    // workspace layout (256B aligned regions)
    char* ws = (char*)d_ws;
    size_t off = 0;
    auto carve = [&](size_t bytes) { size_t o = off; off += (bytes + 255) & ~(size_t)255; return o; };
    float*    dis = (float*)   (ws + carve((size_t)N * sizeof(float)));
    _Float16* A16 = (_Float16*)(ws + carve((size_t)N * DIN * sizeof(_Float16)));
    _Float16* Bpk = (_Float16*)(ws + carve((size_t)DIN * HID * sizeof(_Float16)));
    float*    hw  = (float*)   (ws + carve((size_t)N * HID * sizeof(float)));
    float*    agg = (float*)   (ws + carve((size_t)N * HID * sizeof(float)));
    (void)ws_size; (void)n_in; (void)out_size;

    // ---- degree normalization: dis = rsqrt(deg) ----
    k_init_deg <<<gridFor(N), TPB, 0, stream>>>(dis, N);
    k_count_deg<<<gridFor(E), TPB, 0, stream>>>(dst, dis, E);
    k_to_dis   <<<gridFor(N), TPB, 0, stream>>>(dis, N);

    // ---- layer 1: hw = x @ W1 (WMMA f16) ----
    k_f32_to_f16<<<gridFor((long long)N * DIN), TPB, 0, stream>>>(x, A16, (long long)N * DIN);
    k_pack_B<<<gridFor((long long)DIN * HID), TPB, 0, stream>>>(W1, Bpk, DIN, HID);
    launch_gemm(A16, Bpk, hw, N, DIN, HID, stream);
    k_zero_f32<<<gridFor((long long)N * HID), TPB, 0, stream>>>(agg, (long long)N * HID);
    k_scatter <<<gridFor((long long)E * (HID / 4)), TPB, 0, stream>>>(src, dst, dis, hw, agg, E, HID / 4, HID);
    k_post_relu<<<gridFor((long long)N * HID), TPB, 0, stream>>>(agg, hw, dis, b1, A16, N, HID);

    // ---- layer 2: hw = h1 @ W2 (WMMA f16) ----
    k_pack_B<<<gridFor((long long)HID * OUT), TPB, 0, stream>>>(W2, Bpk, HID, OUT);
    launch_gemm(A16, Bpk, hw, N, HID, OUT, stream);
    k_zero_f32<<<gridFor((long long)N * OUT), TPB, 0, stream>>>(agg, (long long)N * OUT);
    k_scatter <<<gridFor((long long)E * (OUT / 4)), TPB, 0, stream>>>(src, dst, dis, hw, agg, E, OUT / 4, OUT);
    // h2 -> d_out[0 : N*OUT) as f32, and -> A16 as f16 for the classifier
    k_post<<<gridFor((long long)N * OUT), TPB, 0, stream>>>(agg, hw, dis, b2, out, A16, N, OUT);

    // ---- classifier: out = h2 @ Wc + bc (WMMA f16) ----
    k_pack_B<<<gridFor((long long)OUT * OUT), TPB, 0, stream>>>(Wc, Bpk, OUT, OUT);
    launch_gemm(A16, Bpk, hw, N, OUT, OUT, stream);
    k_add_bias<<<gridFor((long long)N * OUT), TPB, 0, stream>>>(hw, bc, out + (long long)N * OUT,
                                                                (long long)N * OUT, OUT);
}
